// NeighborhoodSelfAttention_15255723835593
// MI455X (gfx1250) — matrix operand order repaired and checked by reading, not compiled, not run on hardware
//
#include <hip/hip_runtime.h>
#include <hip/hip_bf16.h>

#define HH 96
#define WW 96
#define SS (HH*WW)     // 9216
#define CC 256
#define NH 8
#define DH 32
#define NKEY 160       // padded key count (7*22=154 -> 160)
#define KROW 22        // max union width of 16 adjacent 7-wide windows

typedef __attribute__((ext_vector_type(16))) __bf16 v16bf;
typedef __attribute__((ext_vector_type(8)))  float  v8f;

union BFrag {
    uint4          u[2];
    unsigned short h[16];
    v16bf          v;
};

// ---- gfx1250 async memory->LDS path (guarded; falls back to sync staging) ----
#if defined(__HIP_DEVICE_COMPILE__) && __has_builtin(__builtin_amdgcn_global_load_async_to_lds_b128)
#define HAVE_ASYNC_LDS 1
#else
#define HAVE_ASYNC_LDS 0
#endif

#if HAVE_ASYNC_LDS
typedef int v4i_vs __attribute__((__vector_size__(16)));
typedef v4i_vs __attribute__((address_space(1)))* gptr_v4i;
typedef v4i_vs __attribute__((address_space(3)))* lptr_v4i;

__device__ __forceinline__ void async_copy16(const void* g, void* l) {
    __builtin_amdgcn_global_load_async_to_lds_b128(
        (gptr_v4i)(g), (lptr_v4i)(l), 0, 0);
}
__device__ __forceinline__ void wait_async0() {
#if __has_builtin(__builtin_amdgcn_s_wait_asynccnt)
    __builtin_amdgcn_s_wait_asynccnt(0);
#else
    asm volatile("s_wait_asynccnt 0x0" ::: "memory");
#endif
}
#endif

__device__ __forceinline__ unsigned short f2bf(float f) {
    unsigned int u = __float_as_uint(f);
    u += 0x7FFFu + ((u >> 16) & 1u);   // round-to-nearest-even
    return (unsigned short)(u >> 16);
}

__device__ __forceinline__ v8f wmma_bf16(v16bf a, v16bf b, v8f c) {
    // D = A(16x32 bf16) * B(32x16 bf16) + C(16x16 f32)
    return __builtin_amdgcn_wmma_f32_16x16x32_bf16(false, a, false, b, (short)0, c, false, false);
}

__device__ __forceinline__ int iclamp(int v, int lo, int hi) {
    return v < lo ? lo : (v > hi ? hi : v);
}

// ---------------- fp32 -> bf16 conversions ----------------
__global__ void k_convert_bf16(const float* __restrict__ in,
                               unsigned short* __restrict__ out, int n) {
    int i = blockIdx.x * 256 + threadIdx.x;
    if (i < n) out[i] = f2bf(in[i]);
}

// x [C][S] f32  ->  xb [S][C] bf16
__global__ void k_pack_x(const float* __restrict__ x,
                         unsigned short* __restrict__ xb) {
    int i = blockIdx.x * 256 + threadIdx.x;   // i over S*C
    int s = i >> 8;
    int c = i & 255;
    xb[i] = f2bf(x[c * SS + s]);
}

// ---------------- QKV projection: qkv[S,768] = xb[S,256] @ wqb[768,256]^T + b ----------------
// Split into Q/K/V per-head buffers [head][S][32] bf16; Q pre-scaled by dh^-0.5.
__global__ __launch_bounds__(32) void k_gemm_qkv(
    const unsigned short* __restrict__ xb,
    const unsigned short* __restrict__ wqb,
    const float* __restrict__ bias,
    unsigned short* __restrict__ Qb,
    unsigned short* __restrict__ Kb,
    unsigned short* __restrict__ Vb) {
    const int lane = threadIdx.x;
    const int s0 = blockIdx.x * 16;
    const int n0 = blockIdx.y * 64;
    const int mr = lane & 15;      // A row / B column / D column
    const int hi = lane >> 4;      // half-wave id
    v8f acc[4] = {};
    const unsigned short* arow = xb + (s0 + mr) * CC;
    for (int k0 = 0; k0 < CC; k0 += 32) {
        BFrag A;
        A.u[0] = *(const uint4*)(arow + k0 + hi * 8);
        A.u[1] = *(const uint4*)(arow + k0 + hi * 8 + 16);
        BFrag B[4];
#pragma unroll
        for (int nt = 0; nt < 4; nt++) {
            const unsigned short* brow = wqb + (n0 + nt * 16 + mr) * CC + k0 + hi * 8;
            B[nt].u[0] = *(const uint4*)brow;
            B[nt].u[1] = *(const uint4*)(brow + 16);
        }
#pragma unroll
        for (int nt = 0; nt < 4; nt++)
            acc[nt] = wmma_bf16(A.v, B[nt].v, acc[nt]);
    }
#pragma unroll
    for (int nt = 0; nt < 4; nt++) {
        int n = n0 + nt * 16 + mr;          // output feature (qkv,head,dh)
        float bv = bias[n];
        int sel  = n >> 8;
        int head = (n >> 5) & 7;
        int d    = n & 31;
        unsigned short* dst = (sel == 0) ? Qb : (sel == 1 ? Kb : Vb);
        float scale = (sel == 0) ? 0.17677669529663687f : 1.0f;  // dh^-0.5 folded into Q
#pragma unroll
        for (int v = 0; v < 8; v++) {
            int s = s0 + v + hi * 8;
            float val = (acc[nt][v] + bv) * scale;
            dst[(head * SS + s) * DH + d] = f2bf(val);
        }
    }
}

// ---------------- Neighborhood attention ----------------
// One wave per (16-query row tile, y, head). 10 WMMAs for QK^T, 10 for P*V.
__global__ __launch_bounds__(32) void k_attn(
    const unsigned short* __restrict__ Qb,
    const unsigned short* __restrict__ Kb,
    const unsigned short* __restrict__ Vb,
    unsigned short* __restrict__ O) {
    __shared__ unsigned short lK[NKEY * DH];   // key tile   [key][dh]
    __shared__ unsigned short lV[NKEY * DH];   // value tile [key][dh]
    __shared__ float          lS[16 * NKEY];   // masked scores
    __shared__ unsigned short lP[16 * NKEY];   // exp(scores-max) bf16
    __shared__ float          lR[16];          // 1/sum per query

    const int lane = threadIdx.x;
    const int x0 = blockIdx.x * 16;
    const int y  = blockIdx.y;
    const int h  = blockIdx.z;
    const int mr = lane & 15;
    const int hi = lane >> 4;

    const int cy   = iclamp(y, 3, HH - 4);
    const int row0 = cy - 3;
    const int col0 = iclamp(x0, 3, WW - 4) - 3;

    // ---- stage K,V tiles (160 keys x 32 dh bf16 each) ----
    const unsigned short* Kg = Kb + (size_t)h * SS * DH;
    const unsigned short* Vg = Vb + (size_t)h * SS * DH;
#pragma unroll
    for (int it = 0; it < 20; it++) {
        int chunk = it * 32 + lane;       // 640 16B chunks per matrix
        int j = chunk >> 2;
        int part = (chunk & 3) * 8;
        int jc = j > 153 ? 153 : j;       // clamp padded keys to a safe address
        int r = jc / KROW;
        int c = jc - r * KROW;
        int cc = col0 + c; if (cc > WW - 1) cc = WW - 1;
        int sidx = (row0 + r) * WW + cc;
#if HAVE_ASYNC_LDS
        async_copy16((const void*)(Kg + sidx * DH + part), (void*)&lK[j * DH + part]);
        async_copy16((const void*)(Vg + sidx * DH + part), (void*)&lV[j * DH + part]);
#else
        *(uint4*)&lK[j * DH + part] = *(const uint4*)(Kg + sidx * DH + part);
        *(uint4*)&lV[j * DH + part] = *(const uint4*)(Vg + sidx * DH + part);
#endif
    }

    // ---- Q fragment, loaded while the async staging is in flight ----
    const unsigned short* qp = Qb + ((size_t)h * SS + (size_t)y * WW + x0 + mr) * DH;
    BFrag QA;
    QA.u[0] = *(const uint4*)(qp + hi * 8);
    QA.u[1] = *(const uint4*)(qp + hi * 8 + 16);

#if HAVE_ASYNC_LDS
    wait_async0();
#endif
    __syncthreads();

    // ---- scores: 10 key tiles of 16 ----
#pragma unroll
    for (int t = 0; t < 10; t++) {
        BFrag KB;
        const unsigned short* kp = &lK[(t * 16 + mr) * DH + hi * 8];
        KB.u[0] = *(const uint4*)kp;
        KB.u[1] = *(const uint4*)(kp + 16);
        v8f c = {};
        c = wmma_bf16(QA.v, KB.v, c);
        int j2 = t * 16 + mr;             // key column owned by this lane
        int r2 = j2 / KROW;
        int c2 = j2 - r2 * KROW;
        int colabs = col0 + c2;
#pragma unroll
        for (int v = 0; v < 8; v++) {
            int mrow = v + hi * 8;
            int cxq = iclamp(x0 + mrow, 3, WW - 4);
            int dlt = colabs - cxq;
            bool valid = (j2 < 154) && (dlt >= -3) && (dlt <= 3);
            lS[mrow * NKEY + j2] = valid ? c[v] : -3.0e38f;
        }
    }
    __syncthreads();

    // ---- softmax, one lane per query ----
    if (lane < 16) {
        float* row = &lS[lane * NKEY];
        float mx = -3.0e38f;
        for (int j = 0; j < NKEY; j++) mx = fmaxf(mx, row[j]);
        float sum = 0.f;
        unsigned short* pr = &lP[lane * NKEY];
        for (int j = 0; j < NKEY; j++) {
            float e = __expf(row[j] - mx);
            sum += e;
            pr[j] = f2bf(e);
        }
        lR[lane] = 1.0f / sum;
    }
    __syncthreads();

    // ---- out = P @ V : N = dh (2 tiles of 16), K = keys (5 tiles of 32) ----
#pragma unroll
    for (int nt = 0; nt < 2; nt++) {
        v8f acc = {};
        int n = nt * 16 + mr;            // dh index owned by this lane
#pragma unroll
        for (int kt = 0; kt < 5; kt++) {
            BFrag PA;
            const unsigned short* pp = &lP[mr * NKEY + kt * 32 + hi * 8];
            PA.u[0] = *(const uint4*)pp;
            PA.u[1] = *(const uint4*)(pp + 16);
            BFrag VB;
            int kbase = kt * 32 + hi * 8;
#pragma unroll
            for (int i = 0; i < 16; i++) VB.h[i] = lV[(kbase + i + (i >= 8 ? 8 : 0)) * DH + n];
            acc = wmma_bf16(PA.v, VB.v, acc);
        }
#pragma unroll
        for (int v = 0; v < 8; v++) {
            int mrow = v + hi * 8;
            int sq = y * WW + x0 + mrow;
            float val = acc[v] * lR[mrow];
            O[sq * CC + h * DH + n] = f2bf(val);
        }
    }
}

// ---------------- Output projection: out[co][s] = ab[s,:] . wob[co,:] + b ----------------
__global__ __launch_bounds__(32) void k_gemm_out(
    const unsigned short* __restrict__ ab,
    const unsigned short* __restrict__ wob,
    const float* __restrict__ bias,
    float* __restrict__ out) {
    const int lane = threadIdx.x;
    const int s0 = blockIdx.x * 16;
    const int n0 = blockIdx.y * 64;
    const int mr = lane & 15;
    const int hi = lane >> 4;
    v8f acc[4] = {};
    const unsigned short* arow = ab + (s0 + mr) * CC;
    for (int k0 = 0; k0 < CC; k0 += 32) {
        BFrag A;
        A.u[0] = *(const uint4*)(arow + k0 + hi * 8);
        A.u[1] = *(const uint4*)(arow + k0 + hi * 8 + 16);
        BFrag B[4];
#pragma unroll
        for (int nt = 0; nt < 4; nt++) {
            const unsigned short* brow = wob + (n0 + nt * 16 + mr) * CC + k0 + hi * 8;
            B[nt].u[0] = *(const uint4*)brow;
            B[nt].u[1] = *(const uint4*)(brow + 16);
        }
#pragma unroll
        for (int nt = 0; nt < 4; nt++)
            acc[nt] = wmma_bf16(A.v, B[nt].v, acc[nt]);
    }
#pragma unroll
    for (int nt = 0; nt < 4; nt++) {
        int n = n0 + nt * 16 + mr;
        float bv = bias[n];
#pragma unroll
        for (int v = 0; v < 8; v++) {
            int s = s0 + v + hi * 8;
            out[(size_t)n * SS + s] = acc[nt][v] + bv;
        }
    }
}

extern "C" void kernel_launch(void* const* d_in, const int* in_sizes, int n_in,
                              void* d_out, int out_size, void* d_ws, size_t ws_size,
                              hipStream_t stream) {
    (void)in_sizes; (void)n_in; (void)out_size; (void)ws_size;
    const float* x     = (const float*)d_in[0];   // [1,256,96,96]
    const float* w_qkv = (const float*)d_in[1];   // [768,256]
    const float* b_qkv = (const float*)d_in[2];   // [768]
    const float* w_out = (const float*)d_in[3];   // [256,256]
    const float* b_out = (const float*)d_in[4];   // [256]
    float* out = (float*)d_out;                   // [1,256,96,96]

    char* ws = (char*)d_ws;
    size_t off = 0;
    auto alloc = [&](size_t bytes) -> void* {
        void* p = ws + off;
        off = (off + bytes + 255) & ~(size_t)255;
        return p;
    };
    unsigned short* xb  = (unsigned short*)alloc((size_t)SS * CC * 2);
    unsigned short* wqb = (unsigned short*)alloc((size_t)768 * CC * 2);
    unsigned short* wob = (unsigned short*)alloc((size_t)CC * CC * 2);
    unsigned short* Qb  = (unsigned short*)alloc((size_t)NH * SS * DH * 2);
    unsigned short* Kb  = (unsigned short*)alloc((size_t)NH * SS * DH * 2);
    unsigned short* Vb  = (unsigned short*)alloc((size_t)NH * SS * DH * 2);
    unsigned short* Ab  = (unsigned short*)alloc((size_t)SS * CC * 2);

    k_convert_bf16<<<(768 * CC + 255) / 256, 256, 0, stream>>>(w_qkv, wqb, 768 * CC);
    k_convert_bf16<<<(CC * CC + 255) / 256, 256, 0, stream>>>(w_out, wob, CC * CC);
    k_pack_x<<<SS * CC / 256, 256, 0, stream>>>(x, xb);
    k_gemm_qkv<<<dim3(SS / 16, 768 / 64), 32, 0, stream>>>(xb, wqb, b_qkv, Qb, Kb, Vb);
    k_attn<<<dim3(WW / 16, HH, NH), 32, 0, stream>>>(Qb, Kb, Vb, Ab);
    k_gemm_out<<<dim3(SS / 16, CC / 64), 32, 0, stream>>>(Ab, wob, b_out, out);
}
